// TensorProductRescale_28913719837196
// MI455X (gfx1250) — matrix operand order repaired and checked by reading, not compiled, not run on hardware
//
#include <hip/hip_runtime.h>

typedef __attribute__((ext_vector_type(16))) _Float16 v16h;
typedef __attribute__((ext_vector_type(8)))  _Float16 v8h;
typedef __attribute__((ext_vector_type(8)))  float    v8f;

#define MUL 256
#define DIM_IN1 1024
#define OUT_DIM 1792
#define THREADS 256
#define ROWS_PER_BLOCK 128        // 4 M-groups x 32 rows
#define LDS_PITCH 264             // 256 + 8 f16 pad: 16B-aligned rows, spread banks
#define SLICE (32 * LDS_PITCH)    // one 32-row K-slice of one weight matrix

#ifndef __has_builtin
#define __has_builtin(x) 0
#endif
#if __has_builtin(__builtin_amdgcn_tensor_load_to_lds)
#define USE_TDM 1
#else
#define USE_TDM 0
#endif

enum { F_X0Y0 = 0, F_DOT = 1, F_X1Y0 = 2, F_X0Y1 = 3, F_CROSS = 4, F_NONE = 5 };

// ---------------- weight conversion: f32 -> f16 (into workspace) -------------
__global__ void tpr_convert_weights_f16(const float* __restrict__ w,
                                        _Float16* __restrict__ w16, int n) {
  int i = blockIdx.x * blockDim.x + threadIdx.x;
  if (i < n) w16[i] = (_Float16)w[i];
}

// ---------------- WMMA wrapper ----------------------------------------------
__device__ __forceinline__ v8f wmma_f16(v16h a, v16h b, v8f c) {
  return __builtin_amdgcn_wmma_f32_16x16x32_f16(false, a, false, b,
                                                (short)0, c, false, false);
}

// ---------------- feature (A-fragment) builders ------------------------------
__device__ __forceinline__ v16h feat_x0(const float* __restrict__ xr, float s, int u0) {
  v16h r;
#pragma unroll
  for (int e = 0; e < 8; ++e) {
    r[e]     = (_Float16)(xr[u0 + e] * s);
    r[e + 8] = (_Float16)(xr[u0 + 16 + e] * s);
  }
  return r;
}

__device__ __forceinline__ v16h feat_x1k(const float* __restrict__ xr, int k,
                                         float s, int u0) {
  v16h r;
#pragma unroll
  for (int e = 0; e < 8; ++e) {
    r[e]     = (_Float16)(xr[MUL + 3 * (u0 + e) + k] * s);
    r[e + 8] = (_Float16)(xr[MUL + 3 * (u0 + 16 + e) + k] * s);
  }
  return r;
}

__device__ __forceinline__ v16h feat_dot(const float* __restrict__ xr,
                                         float s0, float s1, float s2, int u0) {
  v16h r;
#pragma unroll
  for (int e = 0; e < 8; ++e) {
    const float* p = xr + MUL + 3 * (u0 + e);
    const float* q = xr + MUL + 3 * (u0 + 16 + e);
    r[e]     = (_Float16)(p[0] * s0 + p[1] * s1 + p[2] * s2);
    r[e + 8] = (_Float16)(q[0] * s0 + q[1] * s1 + q[2] * s2);
  }
  return r;
}

__device__ __forceinline__ v16h feat_cross(const float* __restrict__ xr,
                                           int i1, int i2, float yA, float yB, int u0) {
  v16h r;
#pragma unroll
  for (int e = 0; e < 8; ++e) {
    const float* p = xr + MUL + 3 * (u0 + e);
    const float* q = xr + MUL + 3 * (u0 + 16 + e);
    r[e]     = (_Float16)(p[i1] * yA - p[i2] * yB);
    r[e + 8] = (_Float16)(q[i1] * yA - q[i2] * yB);
  }
  return r;
}

template <int F, int KIDX>
__device__ __forceinline__ v16h build_feat(const float* __restrict__ xr,
                                           float y0, float y10, float y11, float y12,
                                           int u0) {
  constexpr float inv_sqrt3 = 0.57735026918962576451f;
  constexpr float inv_sqrt2 = 0.70710678118654752440f;
  if constexpr (F == F_X0Y0) {
    return feat_x0(xr, y0, u0);
  } else if constexpr (F == F_DOT) {
    return feat_dot(xr, y10 * inv_sqrt3, y11 * inv_sqrt3, y12 * inv_sqrt3, u0);
  } else if constexpr (F == F_X1Y0) {
    return feat_x1k(xr, KIDX, y0, u0);
  } else if constexpr (F == F_X0Y1) {
    const float s = (KIDX == 0) ? y10 : (KIDX == 1) ? y11 : y12;
    return feat_x0(xr, s, u0);
  } else {  // F_CROSS: c_k = x1[:,i1]*y1[i2] - x1[:,i2]*y1[i1], * 1/sqrt2
    constexpr int i1 = (KIDX + 1) % 3;
    constexpr int i2 = (KIDX + 2) % 3;
    const float yA = ((i2 == 0) ? y10 : (i2 == 1) ? y11 : y12) * inv_sqrt2;
    const float yB = ((i1 == 0) ? y10 : (i1 == 1) ? y11 : y12) * inv_sqrt2;
    return feat_cross(xr, i1, i2, yA, yB, u0);
  }
}

// ---------------- weight slice staging ---------------------------------------
#if USE_TDM
typedef __attribute__((ext_vector_type(4))) unsigned int u32x4;
typedef __attribute__((ext_vector_type(8))) int i32x8;
typedef __attribute__((ext_vector_type(4))) int i32x4;

// Low 32 bits of a generic pointer into LDS = LDS byte offset (ISA flat rules).
__device__ __forceinline__ unsigned lds_off(const _Float16* p) {
  return (unsigned)(unsigned long long)(const void*)p;
}

// TDM: DMA one 32x256 f16 K-slice (rows kb..kb+31) global -> LDS.
// Padding: every 128 DWORDs (one 512B row) insert 4 DWORDs -> LDS_PITCH=264 f16.
__device__ __forceinline__ void tdm_load_slice(const _Float16* __restrict__ W,
                                               int kb, unsigned dst_lds_off) {
  unsigned long long ga = (unsigned long long)(const void*)(W + (size_t)kb * MUL);
  u32x4 g0;
  g0[0] = 1u;                                   // count=1, user descriptor
  g0[1] = dst_lds_off;                          // LDS byte address
  g0[2] = (unsigned)ga;                         // global addr [31:0]
  g0[3] = (unsigned)((ga >> 32) & 0x01FFFFFFull) | (2u << 30);  // [56:32] | type=2
  i32x8 g1 = {};
  g1[0] = (1 << 16)    // data_size = 2 bytes
        | (1 << 20)    // pad_enable
        | (6 << 22)    // pad_interval: 128 DWORDs
        | (3 << 25);   // pad_amount:   4 DWORDs (16 B)
  g1[1] = (256 << 16); // tensor_dim0 = 256 (bits 79:48, low half)
  g1[2] = (256 << 16); // tensor_dim1 = 256 (bits 111:80, low half)
  g1[3] = (256 << 16); // tile_dim0 = 256 (bits 127:112)
  g1[4] = 32;          // tile_dim1 = 32 rows
  g1[5] = 256;         // tensor_dim0_stride = 256 elements
  i32x4 gz = {};
#if __clang_major__ >= 23
  i32x8 gz8 = {};
  __builtin_amdgcn_tensor_load_to_lds(g0, g1, gz, gz, gz8, 0);
#else
  __builtin_amdgcn_tensor_load_to_lds(g0, g1, gz, gz, 0);
#endif
}
#else
__device__ __forceinline__ void stage_slice(const _Float16* __restrict__ W, int kb,
                                            _Float16* __restrict__ dst, int tid) {
#pragma unroll
  for (int i = 0; i < 4; ++i) {
    const int c  = tid + i * THREADS;  // 0..1023
    const int r  = c >> 5;             // 32 chunks per row
    const int cc = c & 31;
    *(v8h*)(dst + r * LDS_PITCH + cc * 8) =
        *(const v8h*)(W + (size_t)(kb + r) * MUL + cc * 8);
  }
}
#endif

// B fragment (32x16): lane L holds column wb + L%16, rows K = 8h+e and 16+8h+e (h = L/16).
__device__ __forceinline__ v16h lds_b_frag(const _Float16* __restrict__ s,
                                           int wb, int lane) {
  const _Float16* q = s + wb + (lane & 15);
  const int h8 = (lane >> 4) * 8;
  v16h r;
#pragma unroll
  for (int e = 0; e < 8; ++e) {
    r[e]     = q[(h8 + e) * LDS_PITCH];
    r[e + 8] = q[(16 + h8 + e) * LDS_PITCH];
  }
  return r;
}

// ---------------- one output chunk: C = featA @ WA (+ featB @ WB) ------------
template <int FA, int FB, int KIDX, int OUTMODE>
__device__ __forceinline__ void run_chunk(
    const _Float16* __restrict__ WA, const _Float16* __restrict__ WB,
    _Float16* __restrict__ sA, _Float16* __restrict__ sB,  // each 2*SLICE
    const float* __restrict__ xr0, const float* __restrict__ xr1,
    float y00, float y100, float y110, float y120,
    float y01, float y101, float y111, float y121,
    long rowBase0, int wcolBase, int wave, int lane, int laneHi, int m16, int tid,
    const float* __restrict__ bias, float scale, long outOff,
    float* __restrict__ out) {
  v8f C0[8] = {};
  v8f C1[8] = {};

#if USE_TDM
  // prologue: issue DMA for slice 0 (buffers are free: previous chunk barriered)
  if (wave == 0) {
    tdm_load_slice(WA, 0, lds_off(sA));
    if constexpr (FB != F_NONE) tdm_load_slice(WB, 0, lds_off(sB));
  }
#endif

  for (int ks = 0; ks < 8; ++ks) {
    const int kb  = ks * 32;
    const int cur = ks & 1;
#if USE_TDM
    if (wave == 0) {
      constexpr short NI = (FB != F_NONE) ? 2 : 1;
      if (ks < 7) {  // overlap: DMA next slice into other buffer during compute
        tdm_load_slice(WA, kb + 32, lds_off(sA + (cur ^ 1) * SLICE));
        if constexpr (FB != F_NONE)
          tdm_load_slice(WB, kb + 32, lds_off(sB + (cur ^ 1) * SLICE));
        __builtin_amdgcn_s_wait_tensorcnt(NI);   // current slice landed
      } else {
        __builtin_amdgcn_s_wait_tensorcnt((short)0);
      }
    }
    __syncthreads();  // current slice visible to all waves
    const _Float16* bufA = sA + cur * SLICE;
    const _Float16* bufB = sB + cur * SLICE;
#else
    __syncthreads();
    stage_slice(WA, kb, sA, tid);
    if constexpr (FB != F_NONE) stage_slice(WB, kb, sB, tid);
    __syncthreads();
    const _Float16* bufA = sA;
    const _Float16* bufB = sB;
#endif

    const int u0 = kb + laneHi * 8;
    {
      v16h A0 = build_feat<FA, KIDX>(xr0, y00, y100, y110, y120, u0);
      v16h A1 = build_feat<FA, KIDX>(xr1, y01, y101, y111, y121, u0);
#pragma unroll
      for (int wt = 0; wt < 8; ++wt) {
        v16h b = lds_b_frag(bufA, wcolBase + wt * 16, lane);
        C0[wt] = wmma_f16(A0, b, C0[wt]);   // B fragment reused by both M tiles
        C1[wt] = wmma_f16(A1, b, C1[wt]);
      }
    }
    if constexpr (FB != F_NONE) {
      v16h A0 = build_feat<FB, KIDX>(xr0, y00, y100, y110, y120, u0);
      v16h A1 = build_feat<FB, KIDX>(xr1, y01, y101, y111, y121, u0);
#pragma unroll
      for (int wt = 0; wt < 8; ++wt) {
        v16h b = lds_b_frag(bufB, wcolBase + wt * 16, lane);
        C0[wt] = wmma_f16(A0, b, C0[wt]);
        C1[wt] = wmma_f16(A1, b, C1[wt]);
      }
    }
#if USE_TDM
    __syncthreads();  // consumed: next iteration's issue may overwrite other buf
#endif
  }

  // epilogue: scale (+bias) and store
#pragma unroll
  for (int wt = 0; wt < 8; ++wt) {
    const int wcol = wcolBase + wt * 16 + m16;
    float bv = 0.0f;
    if constexpr (OUTMODE == 0) bv = bias[wcol];
#pragma unroll
    for (int j = 0; j < 8; ++j) {
      const long m0 = rowBase0 + j + laneHi * 8;
      const long m1 = m0 + 16;
      if constexpr (OUTMODE == 0) {
        out[m0 * OUT_DIM + wcol] = C0[wt][j] * scale + bv;
        out[m1 * OUT_DIM + wcol] = C1[wt][j] * scale + bv;
      } else {
        out[m0 * OUT_DIM + outOff + (long)wcol * 3 + KIDX] = C0[wt][j] * scale;
        out[m1 * OUT_DIM + outOff + (long)wcol * 3 + KIDX] = C1[wt][j] * scale;
      }
    }
  }
}

// ---------------- main fused kernel ------------------------------------------
__global__ __launch_bounds__(THREADS)
void tpr_kernel(const float* __restrict__ x, const float* __restrict__ y,
                const _Float16* __restrict__ w16, const float* __restrict__ bias,
                float* __restrict__ out) {
  __shared__ _Float16 sA[2 * SLICE];
  __shared__ _Float16 sB[2 * SLICE];

  const int tid    = threadIdx.x;
  const int wave   = tid >> 5;
  const int lane   = tid & 31;
  const int laneHi = lane >> 4;
  const int m16    = lane & 15;
  const int waveM  = wave >> 1;        // 0..3 -> which 32-row group
  const int waveW  = wave & 1;         // 0..1 -> which 128-col half
  const int wcolBase = waveW * 128;

  const long rowBase0 = (long)blockIdx.x * ROWS_PER_BLOCK + waveM * 32;
  const long row0 = rowBase0 + m16;        // M-tile 0
  const long row1 = rowBase0 + 16 + m16;   // M-tile 1

  const float* xr0 = x + row0 * DIM_IN1;
  const float* xr1 = x + row1 * DIM_IN1;
  __builtin_prefetch(xr0, 0, 0);
  __builtin_prefetch(xr1, 0, 0);

  const float y00  = y[row0 * 4 + 0], y100 = y[row0 * 4 + 1];
  const float y110 = y[row0 * 4 + 2], y120 = y[row0 * 4 + 3];
  const float y01  = y[row1 * 4 + 0], y101 = y[row1 * 4 + 1];
  const float y111 = y[row1 * 4 + 2], y121 = y[row1 * 4 + 3];

  const _Float16* W0 = w16;
  const _Float16* W1 = w16 + 1 * MUL * MUL;
  const _Float16* W2 = w16 + 2 * MUL * MUL;
  const _Float16* W3 = w16 + 3 * MUL * MUL;
  const _Float16* W4 = w16 + 4 * MUL * MUL;

  const float a2 = 0.044194173824159220275f;  // 1/sqrt(2*MUL)
  const float a1 = 0.0625f;                   // 1/sqrt(MUL)
  const float inv_sqrt_fan = 1.0f / 64.0f;    // 1/sqrt(4096)
  const float scale01 = a2 * inv_sqrt_fan;
  const float scale2  = a1 * inv_sqrt_fan;

  // out0 = (x0*y0) @ W0 + dot @ W3, + bias
  run_chunk<F_X0Y0, F_DOT, 0, 0>(W0, W3, sA, sB, xr0, xr1,
      y00, y100, y110, y120, y01, y101, y111, y121,
      rowBase0, wcolBase, wave, lane, laneHi, m16, tid, bias, scale01, 0, out);

  // out1_k = (x1_k*y0) @ W1 + (x0*y1_k) @ W2
  run_chunk<F_X1Y0, F_X0Y1, 0, 1>(W1, W2, sA, sB, xr0, xr1,
      y00, y100, y110, y120, y01, y101, y111, y121,
      rowBase0, wcolBase, wave, lane, laneHi, m16, tid, bias, scale01, MUL, out);
  run_chunk<F_X1Y0, F_X0Y1, 1, 1>(W1, W2, sA, sB, xr0, xr1,
      y00, y100, y110, y120, y01, y101, y111, y121,
      rowBase0, wcolBase, wave, lane, laneHi, m16, tid, bias, scale01, MUL, out);
  run_chunk<F_X1Y0, F_X0Y1, 2, 1>(W1, W2, sA, sB, xr0, xr1,
      y00, y100, y110, y120, y01, y101, y111, y121,
      rowBase0, wcolBase, wave, lane, laneHi, m16, tid, bias, scale01, MUL, out);

  // out2_k = cross_k(x1,y1)/sqrt2 @ W4
  run_chunk<F_CROSS, F_NONE, 0, 1>(W4, W4, sA, sB, xr0, xr1,
      y00, y100, y110, y120, y01, y101, y111, y121,
      rowBase0, wcolBase, wave, lane, laneHi, m16, tid, bias, scale2, 4 * MUL, out);
  run_chunk<F_CROSS, F_NONE, 1, 1>(W4, W4, sA, sB, xr0, xr1,
      y00, y100, y110, y120, y01, y101, y111, y121,
      rowBase0, wcolBase, wave, lane, laneHi, m16, tid, bias, scale2, 4 * MUL, out);
  run_chunk<F_CROSS, F_NONE, 2, 1>(W4, W4, sA, sB, xr0, xr1,
      y00, y100, y110, y120, y01, y101, y111, y121,
      rowBase0, wcolBase, wave, lane, laneHi, m16, tid, bias, scale2, 4 * MUL, out);
}

// ---------------- launch ------------------------------------------------------
extern "C" void kernel_launch(void* const* d_in, const int* in_sizes, int n_in,
                              void* d_out, int out_size, void* d_ws, size_t ws_size,
                              hipStream_t stream) {
  const float* x    = (const float*)d_in[0];
  const float* y    = (const float*)d_in[1];
  const float* w    = (const float*)d_in[2];
  const float* bias = (const float*)d_in[3];
  float* out = (float*)d_out;

  const int n  = in_sizes[0] / DIM_IN1;  // N = 65536
  const int nw = 5 * MUL * MUL;          // 327680 weight elems (640 KB as f16)
  _Float16* w16 = (_Float16*)d_ws;

  tpr_convert_weights_f16<<<(nw + 255) / 256, 256, 0, stream>>>(w, w16, nw);
  tpr_kernel<<<n / ROWS_PER_BLOCK, THREADS, 0, stream>>>(x, y, w16, bias, out);
}
